// BertAttention_23931557773333
// MI455X (gfx1250) — compile-verified, matrix operand order checked by reading, not run on hardware
//
#include <hip/hip_runtime.h>
#include <hip/hip_bf16.h>

// ---------------------------------------------------------------------------
// BERT attention layer for gfx1250 (MI455X): bf16 WMMA + TDM async staging.
//   B=8, S=1024, H=1024, heads=16, head_dim=64, fp32 I/O.
// ---------------------------------------------------------------------------

typedef __attribute__((ext_vector_type(16))) __bf16 v16bf;
typedef __attribute__((ext_vector_type(8)))  __bf16 v8bf;
typedef __attribute__((ext_vector_type(8)))  float  v8f;

static constexpr int kB  = 8;
static constexpr int kS  = 1024;
static constexpr int kH  = 1024;
static constexpr int kNH = 16;
static constexpr int kHD = 64;
static constexpr int kM  = kB * kS;   // 8192 rows for projections

static constexpr int kKStep = 32;     // k per WMMA step
static constexpr int kMT    = 32;     // rows per wave tile
static constexpr int kNT    = 64;     // cols per wave tile

#if defined(__gfx1250__) && __has_builtin(__builtin_amdgcn_tensor_load_to_lds)
#define HAS_TDM 1
#else
#define HAS_TDM 0
#endif

// float -> bf16, round to nearest even
__device__ __forceinline__ unsigned short f2bf(float f) {
  unsigned int u = __float_as_uint(f);
  u += 0x7FFFu + ((u >> 16) & 1u);
  return (unsigned short)(u >> 16);
}

union ABFrag {
  v16bf v;
  struct { v8bf lo, hi; } h;
};

// Load one 16-bit A/B WMMA fragment for this lane.
// 16-bit matrix 16x32 layout (ISA 7.12.2): lanes 0-15 hold K=[0..7]+[16..23],
// lanes 16-31 hold K=[8..15]+[24..31]. `p` points at (row, kTile); the row is
// contiguous in k.
__device__ __forceinline__ v16bf load_frag(const unsigned short* p, int half) {
  ABFrag f;
  f.h.lo = *(const v8bf*)(p + 8 * half);
  f.h.hi = *(const v8bf*)(p + 16 + 8 * half);
  return f.v;
}

#if HAS_TDM
typedef unsigned int v4u __attribute__((ext_vector_type(4)));
typedef int          v8i __attribute__((ext_vector_type(8)));
typedef int          v4i __attribute__((ext_vector_type(4)));

// One TDM DMA: 2D bf16 tile [tile_rows x 32] from global (row stride 1024
// elements) into LDS, row-major packed. Descriptor per ISA 8.3/8.4:
//   group0: count=1 | lds_addr | global_addr[56:0] | type=2
//   group1: data_size=2B | tensor dims | tile dims | dim0 stride
// This toolchain's builtin takes 6 args (g0, g1, g2, g3, g4, cpol); the extra
// int32x8 group is zero-filled (2D tensor: higher-dim groups unused).
__device__ __forceinline__ void tdm_load_tile(unsigned short* lds_dst,
                                              const unsigned short* gsrc,
                                              int tile_rows) {
  const unsigned long long ga = (unsigned long long)(const void*)gsrc;
  v4u g0;
  g0.x = 1u;                                               // count = 1
  g0.y = (unsigned)(unsigned long long)(void*)lds_dst;     // lds_addr (wave-rel)
  g0.z = (unsigned)ga;                                     // global_addr[31:0]
  g0.w = ((unsigned)(ga >> 32) & 0x01FFFFFFu) | (2u << 30);// addr[56:32]|type=2

  const unsigned dim1 = 0x00100000u;  // generous row count (OOB headroom)
  v8i g1;
  g1[0] = 0x00010000;                                      // data_size = 2 B
  g1[1] = (int)(((unsigned)kH & 0xFFFFu) << 16);           // tensor_dim0 lo16
  g1[2] = (int)((((unsigned)kH >> 16) & 0xFFFFu) |
                ((dim1 & 0xFFFFu) << 16));                 // dim0 hi | dim1 lo
  g1[3] = (int)(((dim1 >> 16) & 0xFFFFu) |
                ((unsigned)kKStep << 16));                 // dim1 hi | tile_dim0
  g1[4] = tile_rows & 0xFFFF;                              // tile_dim1 (dim2=0)
  g1[5] = kH;                                              // dim0_stride lo32
  g1[6] = 0;
  g1[7] = 0;

  const v4i z4 = {0, 0, 0, 0};
  const v8i z8 = {0, 0, 0, 0, 0, 0, 0, 0};
  __builtin_amdgcn_tensor_load_to_lds(g0, g1, z4, z4, z8, 0);
}
#endif

// ---------------------------------------------------------------------------
// fp32 -> bf16 conversion
// ---------------------------------------------------------------------------
__global__ void cvt_bf16(const float* __restrict__ in,
                         unsigned short* __restrict__ out, int n) {
  int i = blockIdx.x * blockDim.x + threadIdx.x;
  if (i < n) out[i] = f2bf(in[i]);
}

// ---------------------------------------------------------------------------
// WMMA GEMM: C[M,1024] = X[M,1024] @ W[1024,1024]^T  (bf16 in, f32 accum)
// One wave per block computes a 32x64 tile (2 A frags x 4 B frags = 8 WMMAs
// per k-step). A/B tiles are staged into LDS by the Tensor Data Mover with
// double buffering under TENSORcnt; fragments come from ds_load_b128.
// mode 1: bf16 out in [B,NH,S,HD] layout (Q, K)
// mode 2: bf16 out in [B,NH,HD,S] layout (V transposed)
// mode 3: fp32 out row-major [M,H] with fused residual add (output projection)
// ---------------------------------------------------------------------------
__global__ void wmma_gemm(const unsigned short* __restrict__ X,
                          const unsigned short* __restrict__ W,
                          unsigned short* __restrict__ outB,
                          float* __restrict__ outF,
                          const float* __restrict__ resid,
                          int mode) {
  const int lane   = threadIdx.x & 31;
  const int half   = lane >> 4;
  const int lidx   = lane & 15;
  const int tilesN = kH / kNT;                 // 16
  const int mTile  = blockIdx.x / tilesN;
  const int nChunk = blockIdx.x % tilesN;
  const int row0   = mTile * kMT;
  const int col0   = nChunk * kNT;

  v8f acc[2][4];
#pragma unroll
  for (int mi = 0; mi < 2; ++mi)
#pragma unroll
    for (int t = 0; t < 4; ++t)
#pragma unroll
      for (int e = 0; e < 8; ++e) acc[mi][t][e] = 0.0f;

#if HAS_TDM
  // Double-buffered LDS tiles fed by the TDM (12 KB of the 320 KB WGP LDS).
  __shared__ __align__(16) unsigned short At[2][kMT * kKStep];  // 2 x 2 KB
  __shared__ __align__(16) unsigned short Bt[2][kNT * kKStep];  // 2 x 4 KB

  const unsigned short* Xs = X + (size_t)row0 * kH;
  const unsigned short* Ws = W + (size_t)col0 * kH;

  // prologue: two tile pairs in flight (TENSORcnt = 4)
  tdm_load_tile(At[0], Xs + 0, kMT);
  tdm_load_tile(Bt[0], Ws + 0, kNT);
  tdm_load_tile(At[1], Xs + kKStep, kMT);
  tdm_load_tile(Bt[1], Ws + kKStep, kNT);

  const int kSteps = kH / kKStep;  // 32
  for (int ks = 0; ks < kSteps; ++ks) {
    // in-order completion: <=2 outstanding means this buffer's pair landed
    if (ks == kSteps - 1) __builtin_amdgcn_s_wait_tensorcnt(0);
    else                  __builtin_amdgcn_s_wait_tensorcnt(2);

    const unsigned short* ap = &At[ks & 1][0];
    const unsigned short* bp = &Bt[ks & 1][0];
    v16bf a[2];
    a[0] = load_frag(ap + (size_t)lidx * kKStep, half);
    a[1] = load_frag(ap + (size_t)(16 + lidx) * kKStep, half);
#pragma unroll
    for (int t = 0; t < 4; ++t) {
      const v16bf b = load_frag(bp + (size_t)(t * 16 + lidx) * kKStep, half);
      acc[0][t] = __builtin_amdgcn_wmma_f32_16x16x32_bf16(
          false, a[0], false, b, (short)0, acc[0][t], false, false);
      acc[1][t] = __builtin_amdgcn_wmma_f32_16x16x32_bf16(
          false, a[1], false, b, (short)0, acc[1][t], false, false);
    }
    const int kNext = (ks + 2) * kKStep;
    if (kNext < kH) {
      // WMMAs above forced ds-waits, so this buffer is safe to overwrite.
      tdm_load_tile(At[ks & 1], Xs + kNext, kMT);
      tdm_load_tile(Bt[ks & 1], Ws + kNext, kNT);
    }
  }
#else
  // Fallback: direct global_load_b128 fragments (round-1 path, 32-row tile).
  for (int k = 0; k < kH; k += kKStep) {
    v16bf a[2];
    a[0] = load_frag(X + (size_t)(row0 + lidx) * kH + k, half);
    a[1] = load_frag(X + (size_t)(row0 + 16 + lidx) * kH + k, half);
#pragma unroll
    for (int t = 0; t < 4; ++t) {
      const v16bf b = load_frag(W + (size_t)(col0 + t * 16 + lidx) * kH + k, half);
      acc[0][t] = __builtin_amdgcn_wmma_f32_16x16x32_bf16(
          false, a[0], false, b, (short)0, acc[0][t], false, false);
      acc[1][t] = __builtin_amdgcn_wmma_f32_16x16x32_bf16(
          false, a[1], false, b, (short)0, acc[1][t], false, false);
    }
  }
#endif

  // C/D layout: element r of lane -> row (r + 8*half), col (lane&15) of tile.
#pragma unroll
  for (int mi = 0; mi < 2; ++mi) {
#pragma unroll
    for (int t = 0; t < 4; ++t) {
#pragma unroll
      for (int r = 0; r < 8; ++r) {
        const int m = row0 + mi * 16 + r + 8 * half;
        const int n = col0 + t * 16 + lidx;
        const float v = acc[mi][t][r];
        if (mode == 3) {
          outF[(size_t)m * kH + n] = v + resid[(size_t)m * kH + n];
        } else {
          const int b = m >> 10, s = m & (kS - 1);
          const int h = n >> 6,  d = n & (kHD - 1);
          size_t idx;
          if (mode == 1) idx = ((size_t)(b * kNH + h) * kS + s) * kHD + d;
          else           idx = ((size_t)(b * kNH + h) * kHD + d) * kS + s;
          outB[idx] = f2bf(v);
        }
      }
    }
  }
}

// ---------------------------------------------------------------------------
// Flash-style attention per (b, head, 16-row q tile): one wave per block.
// scores = Q K^T / 8 + mask, online softmax, O += P V.
// Q,K: [B,NH,S,64] bf16.  Vt: [B,NH,64,S] bf16 (so P@V B-columns are rows).
// Ctx out: [B,S,H] bf16.
// ---------------------------------------------------------------------------
__global__ void wmma_attn(const unsigned short* __restrict__ Qb,
                          const unsigned short* __restrict__ Kb,
                          const unsigned short* __restrict__ Vtb,
                          const float* __restrict__ mask,
                          unsigned short* __restrict__ Ctxb) {
  __shared__ __align__(16) unsigned short Pls[16 * 32];  // P tile, bf16

  const int lane   = threadIdx.x & 31;
  const int half   = lane >> 4;
  const int lidx   = lane & 15;
  const int qTiles = kS / 16;                 // 64
  const int qTile  = blockIdx.x % qTiles;
  const int bh     = blockIdx.x / qTiles;
  const int b      = bh / kNH;
  const int hh     = bh % kNH;

  const unsigned short* Qp = Qb  + ((size_t)bh * kS + qTile * 16) * kHD;
  const unsigned short* Kp = Kb  + (size_t)bh * kS * kHD;
  const unsigned short* Vp = Vtb + (size_t)bh * kHD * kS;
  const float*        mrow = mask + (size_t)b * kS;

  // Q tile A-fragments (k = d dimension, 64 = 2 x 32), kept in registers.
  const v16bf qa0 = load_frag(Qp + (size_t)lidx * kHD + 0,  half);
  const v16bf qa1 = load_frag(Qp + (size_t)lidx * kHD + 32, half);

  v8f O[4];
  float Mr[8], Lr[8];
#pragma unroll
  for (int t = 0; t < 4; ++t)
#pragma unroll
    for (int e = 0; e < 8; ++e) O[t][e] = 0.0f;
#pragma unroll
  for (int r = 0; r < 8; ++r) { Mr[r] = -1e30f; Lr[r] = 0.0f; }

  const float scale = 0.125f;  // 1/sqrt(64)

  for (int kb = 0; kb < kS; kb += 32) {
    // ---- scores tile 16x32: two 16x16 C fragments, each 2 WMMAs over d ----
    v8f c0, c1;
#pragma unroll
    for (int e = 0; e < 8; ++e) { c0[e] = 0.0f; c1[e] = 0.0f; }
    {
      const unsigned short* kr = Kp + (size_t)(kb + lidx) * kHD;
      const v16bf b0 = load_frag(kr + 0,  half);
      const v16bf b1 = load_frag(kr + 32, half);
      c0 = __builtin_amdgcn_wmma_f32_16x16x32_bf16(false, qa0, false, b0,
                                                   (short)0, c0, false, false);
      c0 = __builtin_amdgcn_wmma_f32_16x16x32_bf16(false, qa1, false, b1,
                                                   (short)0, c0, false, false);
    }
    {
      const unsigned short* kr = Kp + (size_t)(kb + 16 + lidx) * kHD;
      const v16bf b0 = load_frag(kr + 0,  half);
      const v16bf b1 = load_frag(kr + 32, half);
      c1 = __builtin_amdgcn_wmma_f32_16x16x32_bf16(false, qa0, false, b0,
                                                   (short)0, c1, false, false);
      c1 = __builtin_amdgcn_wmma_f32_16x16x32_bf16(false, qa1, false, b1,
                                                   (short)0, c1, false, false);
    }

    const float mv0 = mrow[kb + lidx];
    const float mv1 = mrow[kb + 16 + lidx];

    // ---- online softmax (rows live across the 16-lane half-group) ----
    float alpha[8];
#pragma unroll
    for (int r = 0; r < 8; ++r) {
      const float s0 = c0[r] * scale + mv0;
      const float s1 = c1[r] * scale + mv1;
      float mx = fmaxf(s0, s1);
#pragma unroll
      for (int off = 1; off < 16; off <<= 1)
        mx = fmaxf(mx, __shfl_xor(mx, off, 32));
      const float newM = fmaxf(Mr[r], mx);
      const float a  = __expf(Mr[r] - newM);
      const float p0 = __expf(s0 - newM);
      const float p1 = __expf(s1 - newM);
      float ps = p0 + p1;
#pragma unroll
      for (int off = 1; off < 16; off <<= 1)
        ps += __shfl_xor(ps, off, 32);
      Lr[r] = Lr[r] * a + ps;
      Mr[r] = newM;
      alpha[r] = a;
      // stage P tile row-major in LDS (C-frag -> A-frag relayout)
      Pls[(r + 8 * half) * 32 + lidx]      = f2bf(p0);
      Pls[(r + 8 * half) * 32 + 16 + lidx] = f2bf(p1);
    }

    // rescale running output by per-row alpha
#pragma unroll
    for (int t = 0; t < 4; ++t)
#pragma unroll
      for (int r = 0; r < 8; ++r) O[t][r] *= alpha[r];

    __syncthreads();

    // P tile back as an A fragment (row = lane&15, contiguous in k)
    ABFrag pa;
    pa.h.lo = *(const v8bf*)&Pls[lidx * 32 + 8 * half];
    pa.h.hi = *(const v8bf*)&Pls[lidx * 32 + 16 + 8 * half];

    // O(16x64) += P(16x32) @ V(32x64); B column d = Vt row d, contiguous in s
#pragma unroll
    for (int t = 0; t < 4; ++t) {
      const unsigned short* vr = Vp + (size_t)(t * 16 + lidx) * kS + kb;
      const v16bf vb = load_frag(vr, half);
      O[t] = __builtin_amdgcn_wmma_f32_16x16x32_bf16(
          false, pa.v, false, vb, (short)0, O[t], false, false);
    }
    __syncthreads();
  }

  // normalize and write ctx in [B,S,H] layout (heads re-interleaved)
#pragma unroll
  for (int r = 0; r < 8; ++r) {
    const float inv = 1.0f / Lr[r];
    const int s = qTile * 16 + r + 8 * half;
#pragma unroll
    for (int t = 0; t < 4; ++t) {
      const int d = t * 16 + lidx;
      Ctxb[((size_t)b * kS + s) * kH + hh * kHD + d] = f2bf(O[t][r] * inv);
    }
  }
}

// ---------------------------------------------------------------------------
// TF-style LayerNorm over H=1024 per row (residual already fused in GEMM).
// ---------------------------------------------------------------------------
__global__ void ln_fused(const float* __restrict__ X,
                         const float* __restrict__ gamma,
                         const float* __restrict__ beta,
                         float* __restrict__ out) {
  const int row = blockIdx.x;
  const float* x = X + (size_t)row * kH;

  float s = 0.0f, s2 = 0.0f;
  for (int i = threadIdx.x; i < kH; i += blockDim.x) {
    const float v = x[i];
    s += v; s2 += v * v;
  }
  const int lane = threadIdx.x & 31;
  const int wv   = threadIdx.x >> 5;
#pragma unroll
  for (int off = 1; off < 32; off <<= 1) {
    s  += __shfl_xor(s,  off, 32);
    s2 += __shfl_xor(s2, off, 32);
  }
  __shared__ float sh[16];
  __shared__ float stats[2];
  if (lane == 0) { sh[wv] = s; sh[8 + wv] = s2; }
  __syncthreads();
  if (threadIdx.x == 0) {
    float ts = 0.0f, ts2 = 0.0f;
    for (int i = 0; i < 8; ++i) { ts += sh[i]; ts2 += sh[8 + i]; }
    const float mean = ts / (float)kH;
    const float var  = ts2 / (float)kH - mean * mean;
    stats[0] = mean;
    stats[1] = rsqrtf(var + 1e-12f);
  }
  __syncthreads();
  const float mean = stats[0], inv = stats[1];
  for (int i = threadIdx.x; i < kH; i += blockDim.x)
    out[(size_t)row * kH + i] = gamma[i] * ((x[i] - mean) * inv) + beta[i];
}

// ---------------------------------------------------------------------------
// launch
// ---------------------------------------------------------------------------
extern "C" void kernel_launch(void* const* d_in, const int* in_sizes, int n_in,
                              void* d_out, int out_size, void* d_ws,
                              size_t ws_size, hipStream_t stream) {
  (void)in_sizes; (void)n_in; (void)out_size; (void)ws_size;

  const float* hidden = (const float*)d_in[0];
  const float* mask   = (const float*)d_in[1];
  const float* Wq     = (const float*)d_in[2];
  const float* Wk     = (const float*)d_in[3];
  const float* Wv     = (const float*)d_in[4];
  const float* Wo     = (const float*)d_in[5];
  const float* lnw    = (const float*)d_in[6];
  const float* lnb    = (const float*)d_in[7];
  float* out = (float*)d_out;

  char* ws = (char*)d_ws;
  const size_t MB = 1024u * 1024u;
  unsigned short* Xb  = (unsigned short*)(ws + 0);        // 16 MB
  unsigned short* Wqb = (unsigned short*)(ws + 16 * MB);  // 2 MB each
  unsigned short* Wkb = (unsigned short*)(ws + 18 * MB);
  unsigned short* Wvb = (unsigned short*)(ws + 20 * MB);
  unsigned short* Wob = (unsigned short*)(ws + 22 * MB);
  unsigned short* Qb  = (unsigned short*)(ws + 24 * MB);  // 16 MB
  unsigned short* Kb  = (unsigned short*)(ws + 40 * MB);  // 16 MB
  unsigned short* Vtb = (unsigned short*)(ws + 56 * MB);  // 16 MB
  unsigned short* Ctx = (unsigned short*)(ws + 72 * MB);  // 16 MB (total 88 MB)
  float* Proj = (float*)(ws + 24 * MB);  // 32 MB, aliases Qb/Kb (dead by then)

  // 1) fp32 -> bf16
  cvt_bf16<<<(kM * kH + 255) / 256, 256, 0, stream>>>(hidden, Xb, kM * kH);
  cvt_bf16<<<(kH * kH + 255) / 256, 256, 0, stream>>>(Wq, Wqb, kH * kH);
  cvt_bf16<<<(kH * kH + 255) / 256, 256, 0, stream>>>(Wk, Wkb, kH * kH);
  cvt_bf16<<<(kH * kH + 255) / 256, 256, 0, stream>>>(Wv, Wvb, kH * kH);
  cvt_bf16<<<(kH * kH + 255) / 256, 256, 0, stream>>>(Wo, Wob, kH * kH);

  // 2) Q/K/V projections (WMMA, TDM-staged)
  const dim3 gg((kM / kMT) * (kH / kNT));  // 4096 waves
  wmma_gemm<<<gg, 32, 0, stream>>>(Xb, Wqb, Qb,  nullptr, nullptr, 1);
  wmma_gemm<<<gg, 32, 0, stream>>>(Xb, Wkb, Kb,  nullptr, nullptr, 1);
  wmma_gemm<<<gg, 32, 0, stream>>>(Xb, Wvb, Vtb, nullptr, nullptr, 2);

  // 3) attention (WMMA flash)
  wmma_attn<<<kB * kNH * (kS / 16), 32, 0, stream>>>(Qb, Kb, Vtb, mask, Ctx);

  // 4) output projection + residual (WMMA, fp32 out)
  wmma_gemm<<<gg, 32, 0, stream>>>(Ctx, Wob, nullptr, Proj, hidden, 3);

  // 5) LayerNorm
  ln_fused<<<kM, 256, 0, stream>>>(Proj, lnw, lnb, out);
}
